// ProsodyAwareAttention_4501125726777
// MI455X (gfx1250) — compile-verified, matrix operand order checked by reading, not run on hardware
//
#include <hip/hip_runtime.h>
#include <hip/hip_bf16.h>
#include <math.h>

// ---------------------------------------------------------------------------
// ProsodyAwareAttention on MI455X (gfx1250, wave32):
//   q,k,v = X @ W^T + b           (bf16 WMMA, f32 accum, on-the-fly f32->bf16)
//   S = (Q K^T)/32 + am + pw*pm   (bf16 WMMA, TDM double-buffered LDS tiles)
//   P = softmax(S)                (f32 in, bf16 out)
//   ctx = P V                     (bf16 WMMA, TDM A tiles, f32 accum)
// Workgroup tile 128x256, 8 waves as 2(M) x 4(N), per-wave 64x64 = 16 WMMAs/step.
// ---------------------------------------------------------------------------

typedef __attribute__((ext_vector_type(16))) __bf16 v16bf;
typedef __attribute__((ext_vector_type(8)))  __bf16 v8bf;
typedef __attribute__((ext_vector_type(8)))  float  v8f;

#define BM 128
#define BN 256
#define BK 32
#define LDS_K 48         // padded row stride (elements); 96 B keeps 16-B alignment
#define THREADS 256      // 8 waves (wave32)
#define A_TILE (BM * LDS_K)
#define B_TILE (BN * LDS_K)

static_assert(sizeof(__bf16) == 2, "bf16 size");

#if defined(__gfx1250__) && __has_builtin(__builtin_amdgcn_tensor_load_to_lds) && \
    __has_builtin(__builtin_amdgcn_s_wait_tensorcnt)
#define HAVE_TDM 1
#else
#define HAVE_TDM 0
#endif

#if HAVE_TDM
typedef __attribute__((ext_vector_type(4))) unsigned int u32x4;
typedef __attribute__((ext_vector_type(4))) int          i32x4;
typedef __attribute__((ext_vector_type(8))) int          i32x8;

// Generic pointer to shared memory: low 32 bits hold the LDS byte offset.
__device__ __forceinline__ unsigned int lds_off(const void* p) {
  return (unsigned int)(uintptr_t)p;
}

// 2D TDM load: tile_dim0 = 32 bf16 (contiguous, 64 B = 16 DWORDs), tile_dim1 =
// rows, LDS padding 8 DWORDs per row -> 48-element padded LDS stride.
__device__ __forceinline__ void tdm_load_tile(unsigned int lds_addr, const void* gptr,
                                              int tensor_d0, int tensor_d1,
                                              int row_stride_elems, int rows) {
  const unsigned long long ga = (unsigned long long)(uintptr_t)gptr;
  u32x4 g0;
  g0[0] = 1u;                                                  // count=1, user mode
  g0[1] = lds_addr;                                            // lds_addr [63:32]
  g0[2] = (unsigned int)ga;                                    // global_addr lo
  g0[3] = (unsigned int)((ga >> 32) & 0x1FFFFFFu) | (2u << 30); // addr[56:32], type=2
  i32x8 g1;
  g1[0] = (1 << 16)      // data_size = 2 bytes
        | (1 << 20)      // pad_enable
        | (3 << 22)      // pad_interval: 16 DWORDs (one 32-elem bf16 row)
        | (7 << 25);     // pad_amount: 8 DWORDs (16 bf16 elems -> stride 48)
  g1[1] = (tensor_d0 & 0xFFFF) << 16;                           // tensor_dim0 lo16
  g1[2] = ((tensor_d0 >> 16) & 0xFFFF) | ((tensor_d1 & 0xFFFF) << 16);
  g1[3] = ((tensor_d1 >> 16) & 0xFFFF) | (BK << 16);            // tile_dim0 = 32
  g1[4] = rows;                                                 // tile_dim1
  g1[5] = row_stride_elems;                                     // dim0_stride lo32
  g1[6] = 0;
  g1[7] = 0;
  i32x4 z4 = {0, 0, 0, 0};
#if __clang_major__ >= 23
  i32x8 z8 = {0, 0, 0, 0, 0, 0, 0, 0};
  __builtin_amdgcn_tensor_load_to_lds(g0, g1, z4, z4, z8, 0);
#else
  __builtin_amdgcn_tensor_load_to_lds(g0, g1, z4, z4, 0);
#endif
}
#endif  // HAVE_TDM

// Load one 16x32 bf16 WMMA operand fragment from an LDS tile stored [row][k].
// ISA 16-bit A layout: lanes 0-15 (half=0): elems 0..7 -> K 0..7, 8..15 -> K 16..23
//                      lanes 16-31 (half=1): elems 0..7 -> K 8..15, 8..15 -> K 24..31
__device__ __forceinline__ v16bf load_frag(const __bf16* lds, int row, int lane) {
  const int half = lane >> 4;
  const __bf16* base = lds + row * LDS_K;
  v8bf lo = *(const v8bf*)(base + half * 8);
  v8bf hi = *(const v8bf*)(base + 16 + half * 8);
  return __builtin_shufflevector(lo, hi, 0, 1, 2, 3, 4, 5, 6, 7,
                                         8, 9, 10, 11, 12, 13, 14, 15);
}

// One BK=32 step of the per-wave 64x64 tile: 4 A-frags x 4 B-frags = 16 WMMAs.
__device__ __forceinline__ void wmma_step(const __bf16* lds_a, const __bf16* lds_b,
                                          int a_row0, int b_row0, int lane,
                                          v8f acc[4][4]) {
  v16bf af[4];
#pragma unroll
  for (int mt = 0; mt < 4; ++mt)
    af[mt] = load_frag(lds_a, a_row0 + mt * 16 + (lane & 15), lane);
#pragma unroll
  for (int nt = 0; nt < 4; ++nt) {
    v16bf bfr = load_frag(lds_b, b_row0 + nt * 16 + (lane & 15), lane);
#pragma unroll
    for (int mt = 0; mt < 4; ++mt) {
      acc[mt][nt] = __builtin_amdgcn_wmma_f32_16x16x32_bf16(
          false, af[mt], false, bfr, (short)0, acc[mt][nt], false, false);
    }
  }
}

// ---------------------------------------------------------------------------
// Kernel 2 (defined first so the disasm snippet shows the TDM pipeline):
// S[b][q][n] = (sum_h Q[b][q][h]*K[b][n][h])/32 + am + pw*pm
// per batch: M = N = 2048, K = 1024.  B-matrix column n == K row n.
// Tile staging via Tensor Data Mover (double-buffered), TENSORcnt-tracked.
// LDS (dynamic): 2 x (A 128x48 + B 256x48) bf16 = 72 KiB.
// ---------------------------------------------------------------------------
__global__ __launch_bounds__(THREADS)
void scores_gemm_kernel(const __bf16* __restrict__ Q, const __bf16* __restrict__ Km,
                        const float* __restrict__ amask, const float* __restrict__ pmask,
                        const float* __restrict__ pw, float* __restrict__ S) {
  extern __shared__ __bf16 smem[];
  __bf16* const sa = smem;                 // two A buffers: sa + buf*A_TILE
  __bf16* const sb = smem + 2 * A_TILE;    // two B buffers: sb + buf*B_TILE
  const int t = threadIdx.x;
  const int lane = t & 31, wave = t >> 5;
  const int wm = wave >> 2, wn = wave & 3;
  const int b = blockIdx.z;
  const int m0 = blockIdx.x * BM, n0 = blockIdx.y * BN;
  const __bf16* Qb = Q + (size_t)b * 2048 * 1024;
  const __bf16* Kb = Km + (size_t)b * 2048 * 1024;
  const float w = pw[0];

  v8f acc[4][4] = {};

#if HAVE_TDM
  const __bf16* Qt = Qb + (size_t)m0 * 1024;   // tile row start (wave-uniform)
  const __bf16* Kt = Kb + (size_t)n0 * 1024;

  if (wave == 0) {
    tdm_load_tile(lds_off(sa), Qt, 1024, 2048, 1024, BM);
    tdm_load_tile(lds_off(sb), Kt, 1024, 2048, 1024, BN);
    __builtin_amdgcn_s_wait_tensorcnt(0);
  }
  __syncthreads();

  for (int kk = 0; kk < 1024; kk += BK) {
    const int cur = (kk >> 5) & 1;
    const int nxt = cur ^ 1;
    const bool more = (kk + BK) < 1024;
    if (wave == 0 && more) {  // overlap next-tile DMA with current WMMAs
      tdm_load_tile(lds_off(sa + nxt * A_TILE), Qt + kk + BK, 1024, 2048, 1024, BM);
      tdm_load_tile(lds_off(sb + nxt * B_TILE), Kt + kk + BK, 1024, 2048, 1024, BN);
    }
    wmma_step(sa + cur * A_TILE, sb + cur * B_TILE, wm * 64, wn * 64, lane, acc);
    if (wave == 0 && more) __builtin_amdgcn_s_wait_tensorcnt(0);
    __syncthreads();
  }
#else
  const int fr = t >> 1, fc = (t & 1) * 16;
  for (int kk = 0; kk < 1024; kk += BK) {
    {
      const __bf16* src = Qb + (size_t)(m0 + fr) * 1024 + kk + fc;
      __bf16* dst = sa + fr * LDS_K + fc;
      *(v8bf*)(dst) = *(const v8bf*)(src);
      *(v8bf*)(dst + 8) = *(const v8bf*)(src + 8);
    }
    {
      const __bf16* src = Kb + (size_t)(n0 + t) * 1024 + kk;
      __bf16* dst = sb + t * LDS_K;
#pragma unroll
      for (int i = 0; i < 32; i += 8) *(v8bf*)(dst + i) = *(const v8bf*)(src + i);
    }
    __syncthreads();
    wmma_step(sa, sb, wm * 64, wn * 64, lane, acc);
    __syncthreads();
  }
#endif

  const int half = lane >> 4;
#pragma unroll
  for (int mt = 0; mt < 4; ++mt)
#pragma unroll
    for (int nt = 0; nt < 4; ++nt) {
      const int gn = n0 + wn * 64 + nt * 16 + (lane & 15);
#pragma unroll
      for (int v = 0; v < 8; ++v) {
        const int gm = m0 + wm * 64 + mt * 16 + half * 8 + v;
        const size_t idx = (((size_t)b * 2048) + gm) * 2048 + gn;
        S[idx] = acc[mt][nt][v] * 0.03125f + amask[idx] + w * pmask[idx];
      }
    }
}

// ---------------------------------------------------------------------------
// Kernel 1: Out[m][n] = sum_k X[m][k]*W[n][k] + bias[n]   (fp32 in, bf16 out)
// M = 8192 (B*S), N = K = 1024.  B-matrix column n == W row n (no transpose).
// LDS (dynamic): A 128x48 + B 256x48 bf16 = 36 KiB.
// ---------------------------------------------------------------------------
__global__ __launch_bounds__(THREADS)
void qkv_gemm_kernel(const float* __restrict__ X, const float* __restrict__ W,
                     const float* __restrict__ bias, __bf16* __restrict__ Out) {
  extern __shared__ __bf16 smem[];
  __bf16* lds_a = smem;
  __bf16* lds_b = smem + A_TILE;
  const int t = threadIdx.x;
  const int lane = t & 31, wave = t >> 5;
  const int wm = wave >> 2, wn = wave & 3;
  const int m0 = blockIdx.x * BM, n0 = blockIdx.y * BN;
  const int fr = t >> 1, fc = (t & 1) * 16;

  v8f acc[4][4] = {};

  for (int kk = 0; kk < 1024; kk += BK) {
    {  // A tile: X rows, convert f32 -> bf16 (each thread: 16 elems)
      const float* src = X + (size_t)(m0 + fr) * 1024 + kk + fc;
      __bf16* dst = lds_a + fr * LDS_K + fc;
#pragma unroll
      for (int i = 0; i < 16; i += 4) {
        float4 v = *(const float4*)(src + i);
        dst[i + 0] = (__bf16)v.x; dst[i + 1] = (__bf16)v.y;
        dst[i + 2] = (__bf16)v.z; dst[i + 3] = (__bf16)v.w;
      }
      if (kk + BK < 1024) __builtin_prefetch(src + BK, 0, 1);  // global_prefetch_b8
    }
    {  // B tile: W row (n0+t), 32 cols, convert f32 -> bf16
      const float* src = W + (size_t)(n0 + t) * 1024 + kk;
      __bf16* dst = lds_b + t * LDS_K;
#pragma unroll
      for (int i = 0; i < 32; i += 4) {
        float4 v = *(const float4*)(src + i);
        dst[i + 0] = (__bf16)v.x; dst[i + 1] = (__bf16)v.y;
        dst[i + 2] = (__bf16)v.z; dst[i + 3] = (__bf16)v.w;
      }
      if (kk + BK < 1024) __builtin_prefetch(src + BK, 0, 1);
    }
    __syncthreads();
    wmma_step(lds_a, lds_b, wm * 64, wn * 64, lane, acc);
    __syncthreads();
  }

  const int half = lane >> 4;
#pragma unroll
  for (int mt = 0; mt < 4; ++mt)
#pragma unroll
    for (int nt = 0; nt < 4; ++nt) {
      const int gn = n0 + wn * 64 + nt * 16 + (lane & 15);
      const float b = bias[gn];
#pragma unroll
      for (int v = 0; v < 8; ++v) {
        const int gm = m0 + wm * 64 + mt * 16 + half * 8 + v;
        Out[(size_t)gm * 1024 + gn] = (__bf16)(acc[mt][nt][v] + b);
      }
    }
}

// ---------------------------------------------------------------------------
// Kernel 3: row-wise softmax over 2048 elements; f32 in, bf16 out.
// One workgroup (8 waves) per row; 8192 rows.
// ---------------------------------------------------------------------------
__global__ __launch_bounds__(THREADS)
void softmax_kernel(const float* __restrict__ S, __bf16* __restrict__ P) {
  __shared__ float red[THREADS];
  const int t = threadIdx.x;
  const size_t r = blockIdx.x;
  const float* row = S + r * 2048;
  __bf16* orow = P + r * 2048;

  float lmax = -INFINITY;
  for (int i = t; i < 2048; i += THREADS) lmax = fmaxf(lmax, row[i]);
  red[t] = lmax;
  __syncthreads();
  for (int s = THREADS / 2; s > 0; s >>= 1) {
    if (t < s) red[t] = fmaxf(red[t], red[t + s]);
    __syncthreads();
  }
  const float m = red[0];
  __syncthreads();

  float lsum = 0.f;
  for (int i = t; i < 2048; i += THREADS) lsum += __expf(row[i] - m);
  red[t] = lsum;
  __syncthreads();
  for (int s = THREADS / 2; s > 0; s >>= 1) {
    if (t < s) red[t] += red[t + s];
    __syncthreads();
  }
  const float inv = 1.f / red[0];

  for (int i = t; i < 2048; i += THREADS)
    orow[i] = (__bf16)(__expf(row[i] - m) * inv);
}

// ---------------------------------------------------------------------------
// Kernel 4: ctx[b][q][h] = sum_k P[b][q][k] * V[b][k][h]     (f32 out)
// per batch: M = 2048, N = 1024, K = 2048.
// A (P rows) staged by TDM; V tile transposed into LDS [n][k]; both
// double-buffered so one barrier per iteration keeps the pipeline full.
// LDS (dynamic): 2 x (A 128x48 + B 256x48) bf16 = 72 KiB.
// ---------------------------------------------------------------------------
__global__ __launch_bounds__(THREADS)
void context_gemm_kernel(const __bf16* __restrict__ P, const __bf16* __restrict__ V,
                         float* __restrict__ Out) {
  extern __shared__ __bf16 smem[];
  __bf16* const sa = smem;                 // two A buffers: sa + buf*A_TILE
  __bf16* const sb = smem + 2 * A_TILE;    // two B buffers: sb + buf*B_TILE
  const int t = threadIdx.x;
  const int lane = t & 31, wave = t >> 5;
  const int wm = wave >> 2, wn = wave & 3;
  const int b = blockIdx.z;
  const int m0 = blockIdx.x * BM, n0 = blockIdx.y * BN;
  const int kr = t >> 3, nc = (t & 7) * 32;  // V-transpose fill coords
  const int fr = t >> 1, fc = (t & 1) * 16;
  const __bf16* Pb = P + (size_t)b * 2048 * 2048;
  const __bf16* Vb = V + (size_t)b * 2048 * 1024;

  v8f acc[4][4] = {};

  // manual (transposing) fill of the V tile for K-offset kb into buffer `buf`
  auto fill_b = [&](__bf16* buf, int kb) {
    const __bf16* src = Vb + (size_t)(kb + kr) * 1024 + n0 + nc;
#pragma unroll
    for (int c = 0; c < 32; c += 8) {
      v8bf v0 = *(const v8bf*)(src + c);
#pragma unroll
      for (int i = 0; i < 8; ++i) buf[(nc + c + i) * LDS_K + kr] = v0[i];
    }
  };

#if HAVE_TDM
  const __bf16* Pt = Pb + (size_t)m0 * 2048;
  fill_b(sb, 0);
  if (wave == 0) {
    tdm_load_tile(lds_off(sa), Pt, 2048, 2048, 2048, BM);
    __builtin_amdgcn_s_wait_tensorcnt(0);
  }
  __syncthreads();

  for (int kk = 0; kk < 2048; kk += BK) {
    const int cur = (kk >> 5) & 1;
    const int nxt = cur ^ 1;
    const bool more = (kk + BK) < 2048;
    if (more) {
      if (wave == 0)
        tdm_load_tile(lds_off(sa + nxt * A_TILE), Pt + kk + BK, 2048, 2048, 2048, BM);
      fill_b(sb + nxt * B_TILE, kk + BK);
    }
    wmma_step(sa + cur * A_TILE, sb + cur * B_TILE, wm * 64, wn * 64, lane, acc);
    if (wave == 0 && more) __builtin_amdgcn_s_wait_tensorcnt(0);
    __syncthreads();
  }
#else
  for (int kk = 0; kk < 2048; kk += BK) {
    {
      const __bf16* src = Pb + (size_t)(m0 + fr) * 2048 + kk + fc;
      __bf16* dst = sa + fr * LDS_K + fc;
      *(v8bf*)(dst) = *(const v8bf*)(src);
      *(v8bf*)(dst + 8) = *(const v8bf*)(src + 8);
    }
    fill_b(sb, kk);
    __syncthreads();
    wmma_step(sa, sb, wm * 64, wn * 64, lane, acc);
    __syncthreads();
  }
#endif

  const int half = lane >> 4;
#pragma unroll
  for (int mt = 0; mt < 4; ++mt)
#pragma unroll
    for (int nt = 0; nt < 4; ++nt) {
      const int gn = n0 + wn * 64 + nt * 16 + (lane & 15);
#pragma unroll
      for (int v = 0; v < 8; ++v) {
        const int gm = m0 + wm * 64 + mt * 16 + half * 8 + v;
        Out[(((size_t)b * 2048) + gm) * 1024 + gn] = acc[mt][nt][v];
      }
    }
}

// ---------------------------------------------------------------------------
// Workspace layout (bytes):
//   Q bf16  [8192*1024]   @ 0          (16 MiB)
//   K bf16  [8192*1024]   @ 16 MiB
//   V bf16  [8192*1024]   @ 32 MiB
//   S f32   [4*2048*2048] @ 48 MiB     (64 MiB)
//   P bf16  [4*2048*2048] @ 112 MiB    (32 MiB)      total 144 MiB
// ---------------------------------------------------------------------------
extern "C" void kernel_launch(void* const* d_in, const int* in_sizes, int n_in,
                              void* d_out, int out_size, void* d_ws, size_t ws_size,
                              hipStream_t stream) {
  (void)in_sizes; (void)n_in; (void)out_size; (void)ws_size;
  const float* X     = (const float*)d_in[0];
  const float* amask = (const float*)d_in[1];
  const float* pmask = (const float*)d_in[2];
  const float* Wq    = (const float*)d_in[3];
  const float* bq    = (const float*)d_in[4];
  const float* Wk    = (const float*)d_in[5];
  const float* bk    = (const float*)d_in[6];
  const float* Wv    = (const float*)d_in[7];
  const float* bv    = (const float*)d_in[8];
  const float* pw    = (const float*)d_in[9];
  float* out = (float*)d_out;

  char* ws = (char*)d_ws;
  const size_t QKV = (size_t)8192 * 1024 * 2;       // 16 MiB each
  __bf16* Qb = (__bf16*)(ws);
  __bf16* Kb = (__bf16*)(ws + QKV);
  __bf16* Vb = (__bf16*)(ws + 2 * QKV);
  float*  S  = (float*)(ws + 3 * QKV);
  __bf16* P  = (__bf16*)(ws + 3 * QKV + (size_t)4 * 2048 * 2048 * 4);

  const size_t lds_single = (size_t)(A_TILE + B_TILE) * sizeof(__bf16);  // 36 KiB
  const size_t lds_double = 2 * lds_single;                              // 72 KiB

  dim3 blk(THREADS);
  qkv_gemm_kernel<<<dim3(64, 4), blk, lds_single, stream>>>(X, Wq, bq, Qb);
  qkv_gemm_kernel<<<dim3(64, 4), blk, lds_single, stream>>>(X, Wk, bk, Kb);
  qkv_gemm_kernel<<<dim3(64, 4), blk, lds_single, stream>>>(X, Wv, bv, Vb);
  scores_gemm_kernel<<<dim3(16, 8, 4), blk, lds_double, stream>>>(Qb, Kb, amask, pmask, pw, S);
  softmax_kernel<<<dim3(8192), blk, 0, stream>>>(S, P);
  context_gemm_kernel<<<dim3(16, 4, 4), blk, lds_double, stream>>>(P, Vb, out);
}
